// SparseAttention_19421842112584
// MI455X (gfx1250) — matrix operand order repaired and checked by reading, not compiled, not run on hardware
//
#include <hip/hip_runtime.h>

// WMMA fragment types (wave32)
typedef __attribute__((ext_vector_type(2)))  float  v2f;   // fp32 A/B frag (16x16x4)
typedef __attribute__((ext_vector_type(4)))  float  f4;
typedef __attribute__((ext_vector_type(8)))  float  v8f;   // fp32 C/D frag
typedef __bf16 bf16;
typedef __attribute__((ext_vector_type(8)))  __bf16 v8bf;
typedef __attribute__((ext_vector_type(16))) __bf16 v16bf; // bf16 A/B frag (16x16x32)

#define S_LEN   4096
#define D_MODEL 1024
#define NHEAD   16
#define DKH     64
#define WIN     64

// ---------------- Split-bf16 WMMA GEMM: C[M,N] = A[M,K] * B[K,N] (fp32 in/out) ----
// A = Ah + Al, B = Bh + Bl (bf16 hi + bf16 residual); C ≈ AhBh + AhBl + AlBh in f32.
// Block: 256 threads = 8 waves. Tile BM=128 (16 rows/wave), BN=64 (4 n-subtiles), BK=32.
#define BM 128
#define BN 64
#define BK 32
#define AS_STR 40   // bf16 units; row stride 80B (16B-aligned frag chunks, padded)
#define BS_STR 40

// Build a v16bf fragment from two contiguous 8-element chunks (each 16B aligned).
__device__ __forceinline__ v16bf ld_frag2(const bf16* p0, const bf16* p1) {
    v8bf a = *(const v8bf*)p0;
    v8bf b = *(const v8bf*)p1;
    v16bf r;
    #pragma unroll
    for (int e = 0; e < 8; ++e) { r[e] = a[e]; r[e + 8] = b[e]; }
    return r;
}

template<bool QKV_SCATTER>
__global__ __launch_bounds__(256)
void gemm_bf16x3_wmma(const float* __restrict__ A, const float* __restrict__ B,
                      float* __restrict__ Cplain,
                      float* __restrict__ Qp, float* __restrict__ Kp, float* __restrict__ Vp,
                      int M, int N, int K)
{
    __shared__ __align__(16) bf16 As_h[BM * AS_STR];
    __shared__ __align__(16) bf16 As_l[BM * AS_STR];
    __shared__ __align__(16) bf16 Bt_h[BN * BS_STR];   // transposed: [n][k]
    __shared__ __align__(16) bf16 Bt_l[BN * BS_STR];

    const int ntiles = N / BN;
    const int bm0 = (blockIdx.x / ntiles) * BM;
    const int bn0 = (blockIdx.x % ntiles) * BN;
    const int tid  = threadIdx.x;
    const int wave = tid >> 5;
    const int lane = tid & 31;
    const int lm   = lane & 15;
    const int hlf  = lane >> 4;
    const int wrow = wave * 16;

    v8f acc[4] = {};

    for (int k0 = 0; k0 < K; k0 += BK) {
        // ---- stage A tile (128x32 f32 -> hi/lo bf16), 4 float4 per thread ----
        #pragma unroll
        for (int i = 0; i < 4; ++i) {
            int idx = tid + i * 256;           // 0..1023 float4s
            int row = idx >> 3;                // 8 float4 per 32-col row
            int c4  = (idx & 7) << 2;
            f4 v = *(const f4*)(A + (size_t)(bm0 + row) * K + k0 + c4);
            #pragma unroll
            for (int j = 0; j < 4; ++j) {
                float f = v[j];
                bf16 h = (bf16)f;
                bf16 l = (bf16)(f - (float)h);
                As_h[row * AS_STR + c4 + j] = h;
                As_l[row * AS_STR + c4 + j] = l;
            }
        }
        // ---- stage B tile (32x64 f32 -> hi/lo bf16, transposed), 2 float4 per thread ----
        #pragma unroll
        for (int i = 0; i < 2; ++i) {
            int idx = tid + i * 256;           // 0..511 float4s
            int row = idx >> 4;                // 16 float4 per 64-col row
            int c4  = (idx & 15) << 2;
            f4 v = *(const f4*)(B + (size_t)(k0 + row) * N + bn0 + c4);
            #pragma unroll
            for (int j = 0; j < 4; ++j) {
                float f = v[j];
                bf16 h = (bf16)f;
                bf16 l = (bf16)(f - (float)h);
                Bt_h[(c4 + j) * BS_STR + row] = h;
                Bt_l[(c4 + j) * BS_STR + row] = l;
            }
        }
        __syncthreads();

        // ---- compute: per wave, one 16x32 A frag pair, 4 B frag pairs, 12 WMMAs ----
        // A frag (16-bit 16x32): lane half selects K 0-7/16-23 vs 8-15/24-31.
        const bf16* pa_h = &As_h[(wrow + lm) * AS_STR + 8 * hlf];
        const bf16* pa_l = &As_l[(wrow + lm) * AS_STR + 8 * hlf];
        v16bf ah = ld_frag2(pa_h, pa_h + 16);
        v16bf al = ld_frag2(pa_l, pa_l + 16);
        #pragma unroll
        for (int nt = 0; nt < 4; ++nt) {
            // B frag (16-bit 32x16): lane half selects K 0-7/16-23 vs 8-15/24-31 (matches A).
            const bf16* pb_h = &Bt_h[(nt * 16 + lm) * BS_STR + 8 * hlf];
            const bf16* pb_l = &Bt_l[(nt * 16 + lm) * BS_STR + 8 * hlf];
            v16bf bh = ld_frag2(pb_h, pb_h + 16);
            v16bf bl = ld_frag2(pb_l, pb_l + 16);
            acc[nt] = __builtin_amdgcn_wmma_f32_16x16x32_bf16(
                false, ah, false, bh, (short)0, acc[nt], false, false);
            acc[nt] = __builtin_amdgcn_wmma_f32_16x16x32_bf16(
                false, ah, false, bl, (short)0, acc[nt], false, false);
            acc[nt] = __builtin_amdgcn_wmma_f32_16x16x32_bf16(
                false, al, false, bh, (short)0, acc[nt], false, false);
        }
        __syncthreads();
    }

    // Epilogue. C layout: acc[nt][r] = C[m = r + 8*hlf][n = nt*16 + lm] of the wave tile.
    #pragma unroll
    for (int nt = 0; nt < 4; ++nt) {
        const int n = bn0 + nt * 16 + lm;
        #pragma unroll
        for (int r = 0; r < 8; ++r) {
            const int m = bm0 + wrow + r + hlf * 8;
            const float v = acc[nt][r];
            if (QKV_SCATTER) {
                // n in [0,3072): part (Q/K/V), head, dk. Store as [H][S][64].
                const int part = n >> 10;
                const int rem  = n & 1023;
                const int h    = rem >> 6;
                const int dk   = rem & 63;
                float* dst = (part == 0) ? Qp : (part == 1) ? Kp : Vp;
                dst[((size_t)h * S_LEN + m) * DKH + dk] = v;
            } else {
                Cplain[(size_t)m * N + n] = v;
            }
        }
    }
}

// ---------------- Sliding-window causal attention (exact fp32 WMMA) ----------------
// One wave per (head, 16-query tile). q0 is 16-aligned so the window
// [q0-64, q0+15] is exactly 5 key tiles of 16.
__global__ __launch_bounds__(128)
void swa_attn(const float* __restrict__ Q, const float* __restrict__ Kt,
              const float* __restrict__ V, float* __restrict__ Oattn)
{
    // wave-private P buffer: 16 rows x 80 cols, stride 84 (bank-conflict-free frag reads)
    __shared__ __align__(16) float Plds[4][16 * 84];

    const int tid  = threadIdx.x;
    const int wave = tid >> 5;
    const int lane = tid & 31;
    const int lm   = lane & 15;
    const int hlf  = lane >> 4;

    const int g  = blockIdx.x * 4 + wave;
    const int h  = g >> 8;             // 256 query tiles per head
    const int q0 = (g & 255) << 4;

    const float* Qh = Q  + (size_t)h * S_LEN * DKH;
    const float* Kh = Kt + (size_t)h * S_LEN * DKH;
    const float* Vh = V  + (size_t)h * S_LEN * DKH;

    // Q tile as 16 A-fragments (K=4 chunks over DK=64)
    v2f aq[16];
    #pragma unroll
    for (int kk = 0; kk < 16; ++kk) {
        const int d = kk * 4 + hlf * 2;
        const float* p = Qh + (size_t)(q0 + lm) * DKH + d;
        aq[kk].x = p[0];
        aq[kk].y = p[1];
    }

    // ---- scores: S[16][80] = Q @ K^T over 5 key tiles ----
    v8f sc[5] = {};
    #pragma unroll
    for (int t = 0; t < 5; ++t) {
        const int k0g = q0 - WIN + t * 16;   // tile-uniform: whole key tile valid or not
        if (k0g < 0) continue;               // wave-coherent at runtime: EXEC all-1 or all-0
        const float* Kbase = Kh + (size_t)(k0g + lm) * DKH;
        #pragma unroll
        for (int kk = 0; kk < 16; ++kk) {
            const int d = kk * 4 + hlf * 2;
            v2f b;                            // B[dchunk][key]: K^T fragment
            b.x = Kbase[d];
            b.y = Kbase[d + 1];
            sc[t] = __builtin_amdgcn_wmma_f32_16x16x4_f32(
                false, aq[kk], false, b, (short)0, sc[t], false, false);
        }
    }

    // ---- mask + softmax (row m = r + 8*hlf lives in one 16-lane half-group) ----
    const float scale = 0.125f;              // 1/sqrt(64)
    float pv[5][8];
    #pragma unroll
    for (int r = 0; r < 8; ++r) {
        const int i = q0 + r + hlf * 8;
        float mx = -__builtin_inff();
        #pragma unroll
        for (int t = 0; t < 5; ++t) {
            const int j = q0 - WIN + t * 16 + lm;
            const bool valid = (j >= 0) && (j <= i) && (j >= i - WIN);
            const float s = valid ? sc[t][r] * scale : -__builtin_inff();
            pv[t][r] = s;
            mx = fmaxf(mx, s);
        }
        #pragma unroll
        for (int d = 1; d < 16; d <<= 1) mx = fmaxf(mx, __shfl_xor(mx, d, 32));
        float sum = 0.0f;
        #pragma unroll
        for (int t = 0; t < 5; ++t) {
            const float e = __expf(pv[t][r] - mx);   // exp(-inf - mx) = 0
            pv[t][r] = e;
            sum += e;
        }
        #pragma unroll
        for (int d = 1; d < 16; d <<= 1) sum += __shfl_xor(sum, d, 32);
        const float inv = 1.0f / sum;
        #pragma unroll
        for (int t = 0; t < 5; ++t) pv[t][r] *= inv;
    }

    // ---- relayout P (C-frags) -> A-frags through wave-private LDS ----
    float* Pw = &Plds[wave][0];
    #pragma unroll
    for (int r = 0; r < 8; ++r) {
        const int m = r + hlf * 8;
        #pragma unroll
        for (int t = 0; t < 5; ++t)
            Pw[m * 84 + t * 16 + lm] = pv[t][r];
    }
    __syncthreads();   // safety for the cross-lane LDS exchange

    // ---- O[16][64] = P[16][80] @ V[80][64] ----
    v8f oacc[4] = {};
    #pragma unroll
    for (int kk = 0; kk < 20; ++kk) {
        const int kb = kk * 4 + hlf * 2;
        v2f ap;
        ap.x = Pw[lm * 84 + kb];
        ap.y = Pw[lm * 84 + kb + 1];
        const int j0 = q0 - WIN + kb;        // global key row; P is 0 where j<0
        #pragma unroll
        for (int nt = 0; nt < 4; ++nt) {
            v2f b;
            b.x = (j0     >= 0) ? Vh[(size_t)j0       * DKH + nt * 16 + lm] : 0.0f;
            b.y = (j0 + 1 >= 0) ? Vh[(size_t)(j0 + 1) * DKH + nt * 16 + lm] : 0.0f;
            oacc[nt] = __builtin_amdgcn_wmma_f32_16x16x4_f32(
                false, ap, false, b, (short)0, oacc[nt], false, false);
        }
    }

    // Store attn output in [s][h*64 + dk] layout, ready for the output GEMM.
    #pragma unroll
    for (int nt = 0; nt < 4; ++nt) {
        #pragma unroll
        for (int r = 0; r < 8; ++r) {
            const int m = q0 + r + hlf * 8;
            Oattn[(size_t)m * D_MODEL + h * DKH + nt * 16 + lm] = oacc[nt][r];
        }
    }
}

// ---------------- host launch ----------------
extern "C" void kernel_launch(void* const* d_in, const int* in_sizes, int n_in,
                              void* d_out, int out_size, void* d_ws, size_t ws_size,
                              hipStream_t stream)
{
    (void)in_sizes; (void)n_in; (void)out_size; (void)ws_size;
    const float* x    = (const float*)d_in[0];   // [4096, 1024]
    const float* Wqkv = (const float*)d_in[1];   // [1024, 3072]
    const float* Wo   = (const float*)d_in[2];   // [1024, 1024]
    float* out = (float*)d_out;                  // [4096, 1024]

    // Workspace: Q,K,V each [16][4096][64] fp32 (16 MB) + attn_out [4096][1024] (16 MB) = 64 MB
    float* Qp = (float*)d_ws;
    float* Kp = Qp + (size_t)NHEAD * S_LEN * DKH;
    float* Vp = Kp + (size_t)NHEAD * S_LEN * DKH;
    float* Ao = Vp + (size_t)NHEAD * S_LEN * DKH;

    // 1) fused QKV projection (split-bf16 WMMA), scattered to [H][S][64] per tensor
    {
        dim3 grid((S_LEN / BM) * ((3 * D_MODEL) / BN));   // 32 * 48 = 1536
        gemm_bf16x3_wmma<true><<<grid, 256, 0, stream>>>(
            x, Wqkv, nullptr, Qp, Kp, Vp, S_LEN, 3 * D_MODEL, D_MODEL);
    }
    // 2) sliding-window attention: 4096 (head, qtile) waves, 4 waves/block
    {
        dim3 grid((NHEAD * (S_LEN / 16)) / 4);            // 1024
        swa_attn<<<grid, 128, 0, stream>>>(Qp, Kp, Vp, Ao);
    }
    // 3) output projection (split-bf16 WMMA)
    {
        dim3 grid((S_LEN / BM) * (D_MODEL / BN));         // 32 * 16 = 512
        gemm_bf16x3_wmma<false><<<grid, 256, 0, stream>>>(
            Ao, Wo, out, nullptr, nullptr, nullptr, S_LEN, D_MODEL, D_MODEL);
    }
}
